// UNet_small_36807869726742
// MI455X (gfx1250) — matrix operand order repaired and checked by reading, not compiled, not run on hardware
//
#include <hip/hip_runtime.h>
#include <hip/hip_bf16.h>
#include <math.h>

typedef __bf16 bf16_t;
typedef bf16_t v16bf __attribute__((ext_vector_type(16)));
typedef float  v8f   __attribute__((ext_vector_type(8)));

#define BN_EPS 1e-5f
#define BB 32  // batch

__device__ __forceinline__ v8f wmma_bf16(v16bf a, v16bf b, v8f c) {
  return __builtin_amdgcn_wmma_f32_16x16x32_bf16(false, a, false, b, (short)0, c, false, false);
}

__device__ __forceinline__ void pack8(v16bf& a, int base, float4 f0, float4 f1) {
  a[base + 0] = (bf16_t)f0.x; a[base + 1] = (bf16_t)f0.y;
  a[base + 2] = (bf16_t)f0.z; a[base + 3] = (bf16_t)f0.w;
  a[base + 4] = (bf16_t)f1.x; a[base + 5] = (bf16_t)f1.y;
  a[base + 6] = (bf16_t)f1.z; a[base + 7] = (bf16_t)f1.w;
}

// ---- weight convert: f32 (cout x Ktot) -> bf16 (coutPad x Kpad), zero padded
__global__ void k_convert_w(const float* __restrict__ W, bf16_t* __restrict__ dst,
                            int cout, int Ktot, int coutPad, int Kpad) {
  long total = (long)coutPad * Kpad;
  long i = (long)blockIdx.x * 256 + threadIdx.x;
  if (i >= total) return;
  int o = (int)(i / Kpad), k = (int)(i % Kpad);
  float v = (o < cout && k < Ktot) ? W[(long)o * Ktot + k] : 0.f;
  dst[i] = (bf16_t)v;
}

__global__ void k_zero(float* __restrict__ p, int n) {
  int i = blockIdx.x * 256 + threadIdx.x;
  if (i < n) p[i] = 0.f;
}

// ---- gathered conv GEMM: one wave computes a 16x32 output tile.
// y[(b*Nv+n)*cout + o] = sum_k A[bn,k]*W[o,k] + bias[o]
// A[bn, j*cin+c] = x[(b*Nv + no[n*7+j])*cin + c]
template<bool SMALLK>
__global__ __launch_bounds__(256) void k_conv(
    const float* __restrict__ x, const int* __restrict__ no,
    const bf16_t* __restrict__ Wb, const float* __restrict__ bias,
    float* __restrict__ y, float* __restrict__ sums,
    int Nv, int cin, int cout, int Kpad, int Mtiles, int doSums)
{
  __shared__ float ls[64];
  if (threadIdx.x < 64) ls[threadIdx.x] = 0.f;
  __syncthreads();

  const int  lane   = threadIdx.x & 31;
  const int  wave   = threadIdx.x >> 5;
  const int  mTile  = blockIdx.x * 8 + wave;
  const int  nTile  = blockIdx.y;
  const bool hiHalf = (lane >= 16);
  const int  col0   = nTile * 32 + (lane & 15);
  const int  col1   = col0 + 16;
  const bool activeW = (mTile < Mtiles);

  float s0a = 0.f, q0a = 0.f, s1a = 0.f, q1a = 0.f;

  if (activeW) {
    const int mRow = mTile * 16 + (lane & 15);   // A row handled by this lane
    const int b    = mRow / Nv;
    const int n    = mRow - b * Nv;
    const int ksel = hiHalf ? 16 : 0;            // B-fragment K select
    const int s0   = hiHalf ? 16 : 0;            // A-fragment K base
    const bf16_t* wcol0 = Wb + (size_t)col0 * Kpad + ksel;
    const bf16_t* wcol1 = wcol0 + (size_t)16 * Kpad;
    v8f acc0 = {0.f, 0.f, 0.f, 0.f, 0.f, 0.f, 0.f, 0.f};
    v8f acc1 = {0.f, 0.f, 0.f, 0.f, 0.f, 0.f, 0.f, 0.f};

    if (SMALLK) {
      // cin == 4: K total 28, padded to one 32-chunk. kk -> j = kk>>2.
      const int* nb = no + (size_t)n * 7;
      const int n0 = nb[0], n1 = nb[1], n2 = nb[2], n3 = nb[3];
      const int n4 = nb[4], n5 = nb[5], n6 = nb[6];
      const int ja = hiHalf ? n4 : n0;   // halves 0..3
      const int jb = hiHalf ? n5 : n1;   // halves 4..7
      const int jc = hiHalf ? n6 : n2;   // halves 8..11
      const int jd = hiHalf ? -1 : n3;   // halves 12..15 (pad for hi half)
      const float4 f0 = *(const float4*)(x + ((size_t)b * Nv + ja) * 4);
      const float4 f1 = *(const float4*)(x + ((size_t)b * Nv + jb) * 4);
      const float4 f2 = *(const float4*)(x + ((size_t)b * Nv + jc) * 4);
      float4 f3 = make_float4(0.f, 0.f, 0.f, 0.f);
      if (jd >= 0) f3 = *(const float4*)(x + ((size_t)b * Nv + jd) * 4);
      v16bf a;
      pack8(a, 0, f0, f1);
      pack8(a, 8, f2, f3);
      acc0 = wmma_bf16(a, *(const v16bf*)wcol0, acc0);
      acc1 = wmma_bf16(a, *(const v16bf*)wcol1, acc1);
    } else {
      // cin multiple of 32: a 32-chunk never crosses a neighbor boundary.
      const int* nb = no + (size_t)n * 7;
      #pragma unroll
      for (int j = 0; j < 7; ++j) {
        const float*  xr  = x + ((size_t)b * Nv + nb[j]) * cin + s0;
        const bf16_t* wj0 = wcol0 + j * cin;
        const bf16_t* wj1 = wcol1 + j * cin;
        for (int c0 = 0; c0 < cin; c0 += 32) {
          const float4 f0 = *(const float4*)(xr + c0);
          const float4 f1 = *(const float4*)(xr + c0 + 4);
          const float4 f2 = *(const float4*)(xr + c0 + 8);
          const float4 f3 = *(const float4*)(xr + c0 + 12);
          v16bf a;
          pack8(a, 0, f0, f1);
          pack8(a, 8, f2, f3);
          acc0 = wmma_bf16(a, *(const v16bf*)(wj0 + c0), acc0);
          acc1 = wmma_bf16(a, *(const v16bf*)(wj1 + c0), acc1);
        }
      }
    }

    const float bv0 = (col0 < cout) ? bias[col0] : 0.f;
    const float bv1 = (col1 < cout) ? bias[col1] : 0.f;
    const int rbase = mTile * 16 + (hiHalf ? 8 : 0);
    #pragma unroll
    for (int r = 0; r < 8; ++r) {
      const float v0 = acc0[r] + bv0;
      const float v1 = acc1[r] + bv1;
      if (col0 < cout) y[(size_t)(rbase + r) * cout + col0] = v0;
      if (col1 < cout) y[(size_t)(rbase + r) * cout + col1] = v1;
      s0a += v0; q0a += v0 * v0;
      s1a += v1; q1a += v1 * v1;
    }
  }

  if (doSums) {
    // merge half-wave partial column sums, LDS-combine, 1 atomic/col/block
    s0a += __shfl_down(s0a, 16, 32);
    q0a += __shfl_down(q0a, 16, 32);
    s1a += __shfl_down(s1a, 16, 32);
    q1a += __shfl_down(q1a, 16, 32);
    if (activeW && lane < 16) {
      atomicAdd(&ls[lane],      s0a);
      atomicAdd(&ls[16 + lane], q0a);
      atomicAdd(&ls[32 + lane], s1a);
      atomicAdd(&ls[48 + lane], q1a);
    }
    __syncthreads();
    if (threadIdx.x < 16) {
      const int t = (int)threadIdx.x;
      const int cA = nTile * 32 + t;
      const int cB = cA + 16;
      atomicAdd(&sums[cA],        ls[t]);
      atomicAdd(&sums[cout + cA], ls[16 + t]);
      atomicAdd(&sums[cB],        ls[32 + t]);
      atomicAdd(&sums[cout + cB], ls[48 + t]);
    }
  }
}

// ---- dense GEMM for upconv: one wave computes a 16x32 tile.
__global__ __launch_bounds__(256) void k_gemm(
    const float* __restrict__ x, const bf16_t* __restrict__ Wb,
    const float* __restrict__ bias, float* __restrict__ y,
    int K, int nOut, int Mtiles)
{
  const int lane  = threadIdx.x & 31;
  const int wave  = threadIdx.x >> 5;
  const int mTile = blockIdx.x * 8 + wave;
  if (mTile >= Mtiles) return;
  const int  nTile  = blockIdx.y;
  const bool hiHalf = (lane >= 16);
  const int  col0   = nTile * 32 + (lane & 15);
  const int  mRow   = mTile * 16 + (lane & 15);
  const int  ksel   = hiHalf ? 16 : 0;
  const int  s0     = hiHalf ? 16 : 0;
  const float*  xr    = x  + (size_t)mRow * K + s0;
  const bf16_t* wcol0 = Wb + (size_t)col0 * K + ksel;
  const bf16_t* wcol1 = wcol0 + (size_t)16 * K;
  v8f acc0 = {0.f, 0.f, 0.f, 0.f, 0.f, 0.f, 0.f, 0.f};
  v8f acc1 = {0.f, 0.f, 0.f, 0.f, 0.f, 0.f, 0.f, 0.f};
  for (int c0 = 0; c0 < K; c0 += 32) {
    const float4 f0 = *(const float4*)(xr + c0);
    const float4 f1 = *(const float4*)(xr + c0 + 4);
    const float4 f2 = *(const float4*)(xr + c0 + 8);
    const float4 f3 = *(const float4*)(xr + c0 + 12);
    v16bf a;
    pack8(a, 0, f0, f1);
    pack8(a, 8, f2, f3);
    acc0 = wmma_bf16(a, *(const v16bf*)(wcol0 + c0), acc0);
    acc1 = wmma_bf16(a, *(const v16bf*)(wcol1 + c0), acc1);
  }
  const float bv0 = bias[col0];
  const float bv1 = bias[col0 + 16];
  const int rbase = mTile * 16 + (hiHalf ? 8 : 0);
  #pragma unroll
  for (int r = 0; r < 8; ++r) {
    y[(size_t)(rbase + r) * nOut + col0]      = acc0[r] + bv0;
    y[(size_t)(rbase + r) * nOut + col0 + 16] = acc1[r] + bv1;
  }
}

// ---- BN finalize: sums -> (scale, shift)
__global__ void k_finalize(const float* __restrict__ sums, const float* __restrict__ g,
                           const float* __restrict__ beta, float* __restrict__ ss,
                           int cout, float invRows) {
  int o = blockIdx.x * blockDim.x + threadIdx.x;
  if (o >= cout) return;
  float m   = sums[o] * invRows;
  float var = fmaxf(sums[cout + o] * invRows - m * m, 0.f);
  float rstd = rsqrtf(var + BN_EPS);
  float sc = g[o] * rstd;
  ss[o] = sc;
  ss[cout + o] = beta[o] - m * sc;
}

// ---- fused normalize + tanh, writing into (possibly strided) concat dest
__global__ void k_bntanh(const float* __restrict__ y, const float* __restrict__ ss,
                         float* __restrict__ dst, int cout, int stride, int chOff,
                         long total) {
  long i = (long)blockIdx.x * 256 + threadIdx.x;
  if (i >= total) return;
  int o = (int)(i % cout);
  size_t row = (size_t)(i / cout);
  float v = tanhf(y[i] * ss[o] + ss[cout + o]);
  dst[row * stride + chOff + o] = v;
}

// ---- mean-pool of 7 gathered neighbors (input may live strided in a concat buf)
__global__ void k_pool(const float* __restrict__ in, const int* __restrict__ no,
                       float* __restrict__ out, int Nin, int Nout, int cin,
                       int stride, int chOff, long total) {
  long i = (long)blockIdx.x * 256 + threadIdx.x;
  if (i >= total) return;
  int c = (int)(i % cin);
  long t = i / cin;
  int nO = (int)(t % Nout);
  int b  = (int)(t / Nout);
  float acc = 0.f;
  #pragma unroll
  for (int j = 0; j < 7; ++j) {
    int v = no[(size_t)nO * 7 + j];
    acc += in[((size_t)b * Nin + v) * stride + chOff + c];
  }
  out[i] = acc * (1.f / 7.f);
}

// ---- upconv gather/scatter: y viewed as (b, raw*7, cout); top rows + paired-mean rows
__global__ void k_upscatter(const float* __restrict__ yf, const int* __restrict__ top,
                            const int* __restrict__ down, float* __restrict__ dst,
                            int raw, int nout, int cout, int stride, long total) {
  long i = (long)blockIdx.x * 256 + threadIdx.x;
  if (i >= total) return;
  int c = (int)(i % cout);
  long t = i / cout;
  int m = (int)(t % nout);
  int b = (int)(t / nout);
  const int n7c = 7 * cout;
  float val;
  if (m < raw) {
    int p = top[m];
    val = yf[((size_t)b * raw + p / 7) * n7c + (p % 7) * cout + c];
  } else {
    int ii = m - raw;
    int h = (2 * c >= cout) ? 1 : 0;
    int bse = 2 * c - h * cout;
    int p = down[2 * ii + h];
    const float* r = yf + ((size_t)b * raw + p / 7) * n7c + (p % 7) * cout;
    val = 0.5f * (r[bse] + r[bse + 1]);
  }
  dst[((size_t)b * nout + m) * stride + c] = val;
}

extern "C" void kernel_launch(void* const* d_in, const int* in_sizes, int n_in,
                              void* d_out, int out_size, void* d_ws, size_t ws_size,
                              hipStream_t stream) {
  (void)in_sizes; (void)n_in; (void)out_size; (void)ws_size;

  static const int CIN[15]  = {4,32,32,64,64,128,128,256,256,128,128,64,64,32,32};
  static const int COUT[15] = {32,32,64,64,128,128,256,256,128,128,64,64,32,32,36};
  static const int LVL[15]  = {0,0,1,1,2,2,3,3,2,2,1,1,0,0,0};
  static const int NSv[4]   = {10242,2562,642,162};
  static const int URAW[3]  = {162,642,2562};
  static const int UNOUT[3] = {642,2562,10242};
  static const int UCIN[3]  = {256,128,64};
  static const int UCOUT[3] = {128,64,32};

  // -------- unpack inputs (dict order: x, params[convs then ups], idx) --------
  int p = 0;
  const float* x = (const float*)d_in[p++];
  const float *cW[15], *cBias[15], *cG[15], *cBeta[15];
  for (int i = 0; i < 15; ++i) {
    cW[i] = (const float*)d_in[p++];
    cBias[i] = (const float*)d_in[p++];
    if (i != 14) { cG[i] = (const float*)d_in[p++]; cBeta[i] = (const float*)d_in[p++]; }
    else { cG[i] = nullptr; cBeta[i] = nullptr; }
  }
  const float *uW[3], *uBias[3];
  for (int i = 0; i < 3; ++i) { uW[i] = (const float*)d_in[p++]; uBias[i] = (const float*)d_in[p++]; }
  const int* no[4];
  for (int l = 0; l < 4; ++l) no[l] = (const int*)d_in[p++];
  const int *uTop[3], *uDown[3];
  for (int i = 0; i < 3; ++i) { uTop[i] = (const int*)d_in[p++]; uDown[i] = (const int*)d_in[p++]; }

  // -------- carve workspace --------
  char* wbase = (char*)d_ws;
  size_t off = 0;
  auto carve = [&](size_t bytes) -> void* {
    off = (off + 255) & ~(size_t)255;
    void* r = wbase + off; off += bytes; return r;
  };
  bf16_t* wb[15]; int KPAD[15], CPAD[15];
  for (int i = 0; i < 15; ++i) {
    KPAD[i] = (CIN[i] == 4) ? 32 : 7 * CIN[i];
    CPAD[i] = ((COUT[i] + 31) / 32) * 32;   // multiple of 32 for dual-column tiles
    wb[i] = (bf16_t*)carve((size_t)CPAD[i] * KPAD[i] * sizeof(bf16_t));
  }
  bf16_t* uwb[3];
  for (int i = 0; i < 3; ++i)
    uwb[i] = (bf16_t*)carve((size_t)7 * UCOUT[i] * UCIN[i] * sizeof(bf16_t));
  float* sums = (float*)carve(512 * sizeof(float));
  float* ss   = (float*)carve(512 * sizeof(float));
  float* ybuf = (float*)carve((size_t)BB * 2562 * 224 * sizeof(float));   // max GEMM out
  float* hA   = (float*)carve((size_t)BB * 10242 * 32 * sizeof(float));
  float* hB   = (float*)carve((size_t)BB * 10242 * 32 * sizeof(float));
  float* cc0  = (float*)carve((size_t)BB * 10242 * 64 * sizeof(float));   // skip concat L0
  float* cc1  = (float*)carve((size_t)BB * 2562 * 128 * sizeof(float));   // skip concat L1
  float* cc2  = (float*)carve((size_t)BB * 642 * 256 * sizeof(float));    // skip concat L2

  // -------- weight conversion (every call: launch must be deterministic) ------
  for (int i = 0; i < 15; ++i) {
    long tot = (long)CPAD[i] * KPAD[i];
    k_convert_w<<<(int)((tot + 255) / 256), 256, 0, stream>>>(
        cW[i], wb[i], COUT[i], 7 * CIN[i], CPAD[i], KPAD[i]);
  }
  for (int i = 0; i < 3; ++i) {
    long tot = (long)7 * UCOUT[i] * UCIN[i];
    k_convert_w<<<(int)((tot + 255) / 256), 256, 0, stream>>>(
        uW[i], uwb[i], 7 * UCOUT[i], UCIN[i], 7 * UCOUT[i], UCIN[i]);
  }

  // -------- layer helpers --------
  auto conv = [&](int ci, const float* in, float* ydst) {
    int Nv = NSv[LVL[ci]];
    int Mtiles = (BB * Nv) / 16;
    dim3 grid((Mtiles + 7) / 8, CPAD[ci] / 32);
    int doS = (ci != 14) ? 1 : 0;
    if (doS) k_zero<<<(2 * COUT[ci] + 255) / 256, 256, 0, stream>>>(sums, 2 * COUT[ci]);
    if (CIN[ci] == 4)
      k_conv<true><<<grid, 256, 0, stream>>>(in, no[LVL[ci]], wb[ci], cBias[ci], ydst,
                                             sums, Nv, CIN[ci], COUT[ci], KPAD[ci], Mtiles, doS);
    else
      k_conv<false><<<grid, 256, 0, stream>>>(in, no[LVL[ci]], wb[ci], cBias[ci], ydst,
                                              sums, Nv, CIN[ci], COUT[ci], KPAD[ci], Mtiles, doS);
  };
  auto cb = [&](int ci, const float* in, float* dst, int stride, int chOff) {
    conv(ci, in, ybuf);
    int Nv = NSv[LVL[ci]];
    long tot = (long)BB * Nv * COUT[ci];
    k_finalize<<<1, 256, 0, stream>>>(sums, cG[ci], cBeta[ci], ss, COUT[ci],
                                      1.f / (float)(BB * Nv));
    k_bntanh<<<(int)((tot + 255) / 256), 256, 0, stream>>>(ybuf, ss, dst, COUT[ci],
                                                           stride, chOff, tot);
  };
  auto pool = [&](int lvl, const float* in, float* out, int cin, int stride, int chOff) {
    int Nin = NSv[lvl], Nout = NSv[lvl + 1];
    long tot = (long)BB * Nout * cin;
    k_pool<<<(int)((tot + 255) / 256), 256, 0, stream>>>(in, no[lvl], out, Nin, Nout,
                                                         cin, stride, chOff, tot);
  };
  auto upc = [&](int ui, const float* in, float* dst, int stride) {
    int raw = URAW[ui], nout = UNOUT[ui], cin = UCIN[ui], cout = UCOUT[ui];
    int Mtiles = (BB * raw) / 16;
    dim3 grid((Mtiles + 7) / 8, (7 * cout) / 32);
    k_gemm<<<grid, 256, 0, stream>>>(in, uwb[ui], uBias[ui], ybuf, cin, 7 * cout, Mtiles);
    long tot = (long)BB * nout * cout;
    k_upscatter<<<(int)((tot + 255) / 256), 256, 0, stream>>>(
        ybuf, uTop[ui], uDown[ui], dst, raw, nout, cout, stride, tot);
  };

  // -------- network --------
  cb(0,  x,   hA,  32, 0);        // c1_1 -> (B,10242,32)
  cb(1,  hA,  cc0, 64, 32);       // c1_2 -> h1 region of concat0
  pool(0, cc0, hA, 32, 64, 32);   // -> (B,2562,32)
  cb(2,  hA,  hB,  64, 0);        // c2_1
  cb(3,  hB,  cc1, 128, 64);      // c2_2 -> h2 region of concat1
  pool(1, cc1, hA, 64, 128, 64);  // -> (B,642,64)
  cb(4,  hA,  hB,  128, 0);       // c3_1
  cb(5,  hB,  cc2, 256, 128);     // c3_2 -> h3 region of concat2
  pool(2, cc2, hA, 128, 256, 128);// -> (B,162,128)
  cb(6,  hA,  hB,  256, 0);       // c4_1
  cb(7,  hB,  hA,  256, 0);       // c4_2 -> (B,162,256)
  upc(0, hA,  cc2, 256);          // u1 -> concat2[:,:,0:128]
  cb(8,  cc2, hA,  128, 0);       // c7_1
  cb(9,  hA,  hB,  128, 0);       // c7_2
  upc(1, hB,  cc1, 128);          // u2 -> concat1[:,:,0:64]
  cb(10, cc1, hA,  64, 0);        // c8_1
  cb(11, hA,  hB,  64, 0);        // c8_2
  upc(2, hB,  cc0, 64);           // u3 -> concat0[:,:,0:32]
  cb(12, cc0, hA,  32, 0);        // c9_1
  cb(13, hA,  hB,  32, 0);        // c9_2
  conv(14, hB, (float*)d_out);    // c10 -> output, no BN
}
